// IterativeGatedConv2D_19628000543393
// MI455X (gfx1250) — compile-verified, matrix operand order checked by reading, not run on hardware
//
#include <hip/hip_runtime.h>

// Iterative gated circular depthwise 7x7 conv (FFT reference realized in the
// spatial domain) on CDNA5: v_wmma_f32_16x16x32_f16 with precomputed Toeplitz
// B-operands, f16 ping-pong state planes (halves HBM traffic vs f32), and
// async global->LDS halo staging (global_load_async_to_lds_b128 + ASYNCcnt).

#include <stdint.h>

#define Hc 256
#define Wc 256
#define Cc 64
#define Bc 8
#define HWc (Hc * Wc)
#define NITER 8

// LDS tile: 38 rows x 144 cols f16, origin (y0-3, x0-8). 8-aligned column
// origin => every 8-col chunk is 16B-aligned in LDS *and* global memory.
#define TW 144
#define TROWS 38

typedef __attribute__((ext_vector_type(16))) _Float16 v16h;
typedef __attribute__((ext_vector_type(8)))  _Float16 v8h;
typedef __attribute__((ext_vector_type(8)))  float    v8f;

union V16 { v16h v; v8h h[2]; };

// ---------------------------------------------------------------------------
// Toeplitz WMMA B-operands in exact per-lane VGPR layout:
//   T[c][kid][r][lane][slot], 512 f16 per 32x16 matrix.
//   Band shifted for the x0-8 tile origin: T_r[k][n] = wflip[r][k-n-5]
//   for (k-n-5) in [0,7), else 0.  n = lane&15; k slot map
//   {0..7,16..23} (lanes<16) / {8..15,24..31} (lanes>=16).
// ---------------------------------------------------------------------------
__global__ __launch_bounds__(256) void build_toeplitz(const float* __restrict__ gk,
                                                      const float* __restrict__ hk,
                                                      _Float16* __restrict__ T) {
  int c = blockIdx.x;
  const float* w0 = gk + c * 49;
  const float* w1 = hk + c * 49;
  for (int idx = threadIdx.x; idx < 14 * 512; idx += 256) {
    int kid = idx / (7 * 512);
    int rr  = (idx / 512) % 7;
    int l   = (idx & 511) >> 4;
    int s2  = idx & 15;
    int m   = l & 15, hi = l >> 4;
    int k   = (s2 < 8) ? (s2 + (hi ? 8 : 0)) : (s2 + 8 + (hi ? 8 : 0));
    int d   = k - m - 5;
    float v = 0.0f;
    if (d >= 0 && d < 7) {
      const float* w = kid ? w1 : w0;
      v = w[(6 - rr) * 7 + (6 - d)];   // flipped: FFT path = true circular conv
    }
    T[(size_t)(c * 14 + kid * 7 + rr) * 512 + l * 16 + s2] = (_Float16)v;
  }
}

// ---------------------------------------------------------------------------
// (B,H,W,C) f32 -> (B,C,H,W) f16, LDS-tiled so both sides are coalesced.
// ---------------------------------------------------------------------------
__global__ __launch_bounds__(256) void to_planes(const float* __restrict__ in,
                                                 _Float16* __restrict__ out) {
  __shared__ float t[32][65];
  const int spb = HWc / 32;
  int b  = blockIdx.x / spb;
  int s0 = (blockIdx.x % spb) * 32;
  int tid = threadIdx.x;

  int c = tid & 63, sl = tid >> 6;
#pragma unroll
  for (int i = 0; i < 8; ++i) {
    int s = sl + 4 * i;
    t[s][c] = in[(size_t)(b * HWc + s0 + s) * Cc + c];
  }
  __syncthreads();
  int sw = tid & 31, cg2 = tid >> 5;
#pragma unroll
  for (int i = 0; i < 8; ++i) {
    int cc = cg2 + 8 * i;
    out[(size_t)(b * Cc + cc) * HWc + s0 + sw] = (_Float16)t[sw][cc];
  }
}

// ---------------------------------------------------------------------------
// (B,C,H,W) f16 -> (B,H,W,C) f32
// ---------------------------------------------------------------------------
__global__ __launch_bounds__(256) void from_planes(const _Float16* __restrict__ in,
                                                   float* __restrict__ out) {
  __shared__ float t[32][65];
  const int spb = HWc / 32;
  int b  = blockIdx.x / spb;
  int s0 = (blockIdx.x % spb) * 32;
  int tid = threadIdx.x;

  int sw = tid & 31, cg2 = tid >> 5;
#pragma unroll
  for (int i = 0; i < 8; ++i) {
    int cc = cg2 + 8 * i;
    t[sw][cc] = (float)in[(size_t)(b * Cc + cc) * HWc + s0 + sw];
  }
  __syncthreads();
  int c = tid & 63, sl = tid >> 6;
#pragma unroll
  for (int i = 0; i < 8; ++i) {
    int s = sl + 4 * i;
    out[(size_t)(b * HWc + s0 + s) * Cc + c] = t[s][c];
  }
}

// ---------------------------------------------------------------------------
// One gated-conv iteration. Grid: (B*C) planes x 8 row-tiles x 2 col-tiles.
// Block = 256 threads = 8 waves; each wave owns TWO stacked 16x16 tiles.
// Halo staging: async b128 copies (global f16 -> LDS f16), 16B aligned on
// both sides, circular wrap handled per 8-column chunk (no mid-chunk wrap).
// Inner loop per r: 4x ds_load_b128 + 4 WMMAs; 14 Toeplitz operands live in
// registers. Blend reads h from the staged LDS tile (no global re-read).
// ---------------------------------------------------------------------------
__global__ __launch_bounds__(256) void step_kernel(const _Float16* __restrict__ hin,
                                                   _Float16* __restrict__ hout,
                                                   const _Float16* __restrict__ Tb) {
  __shared__ __align__(16) _Float16 smem[TROWS * TW];

  int bid   = blockIdx.x;
  int plane = bid >> 4;                 // b*C + c
  int rem   = bid & 15;
  int ty    = rem >> 1, tx = rem & 1;
  int y0    = ty * 32, x0 = tx * 128;
  int c     = plane & (Cc - 1);
  int tid   = threadIdx.x;
  int lane  = tid & 31;
  int wv    = tid >> 5;
  int m     = lane & 15;
  int hi    = lane >> 4;

  const _Float16* __restrict__ base = hin + (size_t)plane * HWc;

  // Preload all 14 Toeplitz operands (gate r=0..6, hidden r=0..6) into VGPRs.
  const v16h* __restrict__ Tg = (const v16h*)(Tb + (size_t)c * 14 * 512);
  const v16h* __restrict__ Th = Tg + 7 * 32;
  v16h tg[7], th[7];
#pragma unroll
  for (int r = 0; r < 7; ++r) {
    tg[r] = Tg[r * 32 + lane];
    th[r] = Th[r * 32 + lane];
  }

  // Async-stage the halo tile (38 x 144 f16, circularly wrapped) into LDS.
  // 18 chunks of 8 f16 per row; chunk starts are 8-col aligned -> 16B aligned
  // in both global and LDS, and never straddle the circular wrap.
  {
    uint32_t lds0 = (uint32_t)(uintptr_t)smem;
    const int nchunk = TROWS * (TW / 8);                  // 684
    for (int idx = tid; idx < nchunk; idx += 256) {
      int row = idx / (TW / 8);
      int ch  = idx - row * (TW / 8);
      int gy  = (y0 - 3 + row) & (Hc - 1);
      int gx  = (x0 - 8 + ch * 8) & (Wc - 1);
      uint32_t goff = (uint32_t)(gy * Wc + gx) * 2u;      // byte offset
      uint32_t ldsa = lds0 + (uint32_t)(row * TW + ch * 8) * 2u;
      asm volatile("global_load_async_to_lds_b128 %0, %1, %2"
                   :
                   : "v"(ldsa), "v"(goff), "s"(base)
                   : "memory");
    }
    asm volatile("s_wait_asynccnt 0" ::: "memory");
  }
  __syncthreads();

  v8f cg0 = {}, ch0 = {}, cg1 = {}, ch1 = {};

#pragma unroll
  for (int r = 0; r < 7; ++r) {
    // A tiles: 16x32 f16 slabs at LDS col 16*wv (16B aligned).
    // Lane K layout {0..7,16..23} (lanes<16) / {8..15,24..31} (lanes>=16).
    V16 a0, a1;
    int ro0 = (r + m) * TW + 16 * wv + (hi ? 8 : 0);
    a0.h[0] = *(const v8h*)&smem[ro0];
    a0.h[1] = *(const v8h*)&smem[ro0 + 16];
    int ro1 = ro0 + 16 * TW;
    a1.h[0] = *(const v8h*)&smem[ro1];
    a1.h[1] = *(const v8h*)&smem[ro1 + 16];

    cg0 = __builtin_amdgcn_wmma_f32_16x16x32_f16(false, a0.v, false, tg[r],
                                                 (short)0, cg0, false, false);
    ch0 = __builtin_amdgcn_wmma_f32_16x16x32_f16(false, a0.v, false, th[r],
                                                 (short)0, ch0, false, false);
    cg1 = __builtin_amdgcn_wmma_f32_16x16x32_f16(false, a1.v, false, tg[r],
                                                 (short)0, cg1, false, false);
    ch1 = __builtin_amdgcn_wmma_f32_16x16x32_f16(false, a1.v, false, th[r],
                                                 (short)0, ch1, false, false);
  }

  // Gated blend (f32 math, f16 store). C/D: VGPR j -> row M=j+8*hi, col N=m.
  // h comes from the staged LDS tile: global (y,x) -> LDS (y-y0+3, x-x0+8).
  _Float16* __restrict__ obase = hout + (size_t)plane * HWc;
  int x    = x0 + 16 * wv + m;
  int lcol = 16 * wv + m + 8;
#pragma unroll
  for (int j = 0; j < 8; ++j) {
    int rowi = j + 8 * hi;
    float hv = (float)smem[(3 + rowi) * TW + lcol];
    float z  = 1.0f / (1.0f + __expf(-cg0[j]));
    obase[(size_t)(y0 + rowi) * Wc + x] = (_Float16)((1.0f - z) * hv + z * ch0[j]);
  }
#pragma unroll
  for (int j = 0; j < 8; ++j) {
    int rowi = 16 + j + 8 * hi;
    float hv = (float)smem[(3 + rowi) * TW + lcol];
    float z  = 1.0f / (1.0f + __expf(-cg1[j]));
    obase[(size_t)(y0 + rowi) * Wc + x] = (_Float16)((1.0f - z) * hv + z * ch1[j]);
  }
}

// ---------------------------------------------------------------------------
extern "C" void kernel_launch(void* const* d_in, const int* in_sizes, int n_in,
                              void* d_out, int out_size, void* d_ws, size_t ws_size,
                              hipStream_t stream) {
  (void)in_sizes; (void)n_in; (void)out_size; (void)ws_size;

  const float* x  = (const float*)d_in[0];
  const float* gk = (const float*)d_in[1];
  const float* hk = (const float*)d_in[2];
  float* out = (float*)d_out;

  _Float16* Tbuf = (_Float16*)d_ws;                         // 896 KB
  _Float16* buf0 = (_Float16*)((char*)d_ws + (1 << 20));    // 67 MB f16 planes
  _Float16* buf1 = (_Float16*)d_out;                        // reuse d_out (134 MB)

  const int tBlocks = Bc * (HWc / 32);   // 16384
  const int sBlocks = Bc * Cc * 16;      // 8192

  build_toeplitz<<<Cc, 256, 0, stream>>>(gk, hk, Tbuf);
  to_planes<<<tBlocks, 256, 0, stream>>>(x, buf0);

  _Float16* src = buf0;
  _Float16* dst = buf1;
  for (int it = 0; it < NITER; ++it) {
    step_kernel<<<sBlocks, 256, 0, stream>>>(src, dst, Tbuf);
    _Float16* t = src; src = dst; dst = t;
  }
  // NITER even -> final state back in buf0 (== src); d_out free to overwrite.
  from_planes<<<tBlocks, 256, 0, stream>>>(src, out);
}
